// DualModel_61117384622159
// MI455X (gfx1250) — compile-verified, matrix operand order checked by reading, not run on hardware
//
#include <hip/hip_runtime.h>
#include <hip/hip_bf16.h>

typedef __attribute__((ext_vector_type(16))) __bf16 v16bf;
typedef __attribute__((ext_vector_type(8)))  __bf16 v8bf;
typedef __attribute__((ext_vector_type(8)))  float  v8f;

#define BATCH   2048
#define BK      32
#define LDSROW  40   // 32 data + 8 pad bf16 -> 80B row stride, conflict-free b128 reads

// per-buffer LDS geometry (elements)
#define AMAT_ELT   (128 * LDSROW)          // one 128x32 matrix (padded)
#define AREG_ELT   (3 * AMAT_ELT)          // 3 A matrices
#define BUF_ELT    (6 * AMAT_ELT)          // 3 A + 3 B matrices
// total LDS: 2 * BUF_ELT * 2B = 122880 B

// ---------------------------------------------------------------------------
// Elementwise prep kernels
// ---------------------------------------------------------------------------

__global__ __launch_bounds__(256)
void prep_w_kernel(const float* __restrict__ w, const float* __restrict__ wa,
                   const float* __restrict__ we, const float* __restrict__ wv,
                   __bf16* __restrict__ mid, __bf16* __restrict__ eps,
                   __bf16* __restrict__ ab, int n) {
    int i = blockIdx.x * 256 + threadIdx.x;
    if (i >= n) return;
    float W = w[i];
    float inter = fminf(wa[i] + we[i] - W, W - (wa[i] - we[i]));
    float e = inter / (1.f + __expf(-wv[i]));
    mid[i] = (__bf16)W;
    eps[i] = (__bf16)e;
    ab[i]  = (__bf16)(fabsf(W) + e);
}

__global__ __launch_bounds__(256)
void prep_b_kernel(const float* __restrict__ b, const float* __restrict__ ba,
                   const float* __restrict__ be, const float* __restrict__ bv,
                   float* __restrict__ blo, float* __restrict__ bhi, int n) {
    int i = blockIdx.x * 256 + threadIdx.x;
    if (i >= n) return;
    float B = b[i];
    float inter = fminf(ba[i] + be[i] - B, B - (ba[i] - be[i]));
    float e = inter / (1.f + __expf(-bv[i]));
    blo[i] = B - e;
    bhi[i] = B + e;
}

__global__ __launch_bounds__(256)
void prep_x_kernel(const float* __restrict__ x,
                   __bf16* __restrict__ mu, __bf16* __restrict__ ab,
                   __bf16* __restrict__ r, int n) {
    int i = blockIdx.x * 256 + threadIdx.x;
    if (i >= n) return;
    float v = x[i];
    mu[i] = (__bf16)v;
    ab[i] = (__bf16)fabsf(v);
    r[i]  = (__bf16)0.0f;
}

// ---------------------------------------------------------------------------
// CDNA5 async copy: global -> LDS, 16B per lane, ASYNCcnt-tracked.
// ---------------------------------------------------------------------------
__device__ __forceinline__ void async_copy_b128(unsigned lds_addr, const __bf16* g) {
    asm volatile("global_load_async_to_lds_b128 %0, %1, off"
                 :: "v"(lds_addr), "v"((unsigned long long)(uintptr_t)g)
                 : "memory");
}
__device__ __forceinline__ void wait_async0() {
    asm volatile("s_wait_asynccnt 0x0" ::: "memory");
}

__device__ __forceinline__ v16bf load_frag(const __bf16* rowbase, int kb) {
    v8bf lo = *(const v8bf*)(rowbase + kb);        // K = kb .. kb+7
    v8bf hi = *(const v8bf*)(rowbase + kb + 16);   // K = kb+16 .. kb+23
    return __builtin_shufflevector(lo, hi, 0,1,2,3,4,5,6,7,8,9,10,11,12,13,14,15);
}

// ---------------------------------------------------------------------------
// Fused interval GEMM: H = mu_A@w^T ; K = |mu_A|@eps^T + r_A@(|w|+eps)^T
// Block 256 thr (8 wave32), tile 128(M) x 128(N) x 32(K), double-buffered LDS
// fed by global_load_async_to_lds_b128. Wave tile 32x64 = 2x4 WMMA tiles with
// two f32 accumulators (H,K): 24 v_wmma per k-step per wave.
// A fragments (3x2 = 48 VGPRs) are held and reused across the 4 n-tiles; B
// fragments are streamed 3-at-a-time to keep live VGPRs < 256 (no MSB churn).
// B matrices are [N][K] (natural [dout][din] weight layout).
// ---------------------------------------------------------------------------
__global__ __launch_bounds__(256)
void ival_gemm_kernel(const __bf16* __restrict__ Amu, const __bf16* __restrict__ Aab,
                      const __bf16* __restrict__ Ar,
                      const __bf16* __restrict__ Bmid, const __bf16* __restrict__ Beps,
                      const __bf16* __restrict__ Bab,
                      const float* __restrict__ blo, const float* __restrict__ bhi,
                      __bf16* __restrict__ Omu, __bf16* __restrict__ Oab,
                      __bf16* __restrict__ Or, float* __restrict__ Out,
                      int M, int N, int K, int relu_next) {
    __shared__ __bf16 sh[2 * BUF_ELT];

    const int t    = threadIdx.x;
    const int lane = t & 31;
    const int wave = t >> 5;
    const int wm   = wave >> 1;        // 0..3  (32 rows each)
    const int wn   = wave & 1;         // 0..1  (64 cols each)
    const int lrow = lane & 15;
    const int kb   = (lane >> 4) << 3; // 0 or 8

    const int m0 = blockIdx.y * 128;
    const int n0 = blockIdx.x * 128;

    const __bf16* Ap[3] = {Amu, Aab, Ar};
    const __bf16* Bp[3] = {Bmid, Beps, Bab};

    const unsigned shbase = (unsigned)(uintptr_t)(void*)&sh[0];

    // async stage of one 128x32 k-tile for all 6 matrices into buffer `buf`:
    // per matrix: 128 rows x 4 b128 chunks = 512 chunks; 6 matrices = 3072
    // chunks; 256 threads -> 12 async loads per thread.
    auto stage = [&](int buf, int k0) {
        const unsigned base = shbase + (unsigned)buf * (BUF_ELT * 2);
#pragma unroll
        for (int j = 0; j < 6; ++j) {          // A matrices
            int idx = t + j * 256;
            int mat = idx >> 9, r = (idx >> 2) & 127, cc = (idx & 3) * 8;
            async_copy_b128(base + (unsigned)(((mat * 128 + r) * LDSROW + cc) * 2),
                            Ap[mat] + (size_t)(m0 + r) * K + k0 + cc);
        }
#pragma unroll
        for (int j = 0; j < 6; ++j) {          // B matrices
            int idx = t + j * 256;
            int mat = idx >> 9, r = (idx >> 2) & 127, cc = (idx & 3) * 8;
            async_copy_b128(base + (unsigned)((AREG_ELT + (mat * 128 + r) * LDSROW + cc) * 2),
                            Bp[mat] + (size_t)(n0 + r) * K + k0 + cc);
        }
    };

    const v8f vzero = {0.f, 0.f, 0.f, 0.f, 0.f, 0.f, 0.f, 0.f};
    v8f accH[2][4], accK[2][4];
#pragma unroll
    for (int mt = 0; mt < 2; ++mt)
#pragma unroll
        for (int nt = 0; nt < 4; ++nt) { accH[mt][nt] = vzero; accK[mt][nt] = vzero; }

    stage(0, 0);                                // prologue: fill buffer 0

    int buf = 0;
    for (int k0 = 0; k0 < K; k0 += BK) {
        wait_async0();                          // this wave's async writes done
        __syncthreads();                        // => every wave's writes done
        if (k0 + BK < K) stage(buf ^ 1, k0 + BK);

        const __bf16* lA = &sh[buf * BUF_ELT];
        const __bf16* lB = lA + AREG_ELT;

        // A fragments, held across all 4 n-tiles: 3 matrices x 2 m-tiles
        v16bf af[3][2];
#pragma unroll
        for (int mt = 0; mt < 2; ++mt) {
            const int arow = wm * 32 + mt * 16 + lrow;
#pragma unroll
            for (int mat = 0; mat < 3; ++mat)
                af[mat][mt] = load_frag(&lA[(mat * 128 + arow) * LDSROW], kb);
        }

#pragma unroll
        for (int nt = 0; nt < 4; ++nt) {
            const int brow = wn * 64 + nt * 16 + lrow;
            v16bf b0 = load_frag(&lB[(0 * 128 + brow) * LDSROW], kb);
            v16bf b1 = load_frag(&lB[(1 * 128 + brow) * LDSROW], kb);
            v16bf b2 = load_frag(&lB[(2 * 128 + brow) * LDSROW], kb);
#pragma unroll
            for (int mt = 0; mt < 2; ++mt) {
                accH[mt][nt] = __builtin_amdgcn_wmma_f32_16x16x32_bf16(
                    false, af[0][mt], false, b0, (short)0, accH[mt][nt], false, false);
                accK[mt][nt] = __builtin_amdgcn_wmma_f32_16x16x32_bf16(
                    false, af[1][mt], false, b1, (short)0, accK[mt][nt], false, false);
                accK[mt][nt] = __builtin_amdgcn_wmma_f32_16x16x32_bf16(
                    false, af[2][mt], false, b2, (short)0, accK[mt][nt], false, false);
            }
        }
        buf ^= 1;
    }

    // --- epilogue: bias, (ReLU), fused next-layer operand conversion ---
    const int rhalf = (lane >> 4) << 3;  // C/D layout: lanes 16..31 hold M+8
#pragma unroll
    for (int mt = 0; mt < 2; ++mt) {
#pragma unroll
        for (int nt = 0; nt < 4; ++nt) {
            const int c  = n0 + wn * 64 + nt * 16 + lrow;
            const float bl = blo[c], bh = bhi[c];
            const int rbase = m0 + wm * 32 + mt * 16 + rhalf;
            v8f H = accH[mt][nt], Kv = accK[mt][nt];
#pragma unroll
            for (int v = 0; v < 8; ++v) {
                const int rr = rbase + v;
                float lo = H[v] - Kv[v] + bl;
                float up = H[v] + Kv[v] + bh;
                if (relu_next) {
                    lo = fmaxf(lo, 0.f);
                    up = fmaxf(up, 0.f);
                    const float mu = 0.5f * (lo + up);
                    const float rd = 0.5f * (up - lo);
                    const size_t o = (size_t)rr * N + c;
                    Omu[o] = (__bf16)mu;
                    Oab[o] = (__bf16)fabsf(mu);
                    Or[o]  = (__bf16)rd;
                } else {
                    const size_t o = ((size_t)rr * N + c) * 2;
                    Out[o + 0] = lo;
                    Out[o + 1] = up;
                }
            }
        }
    }
}

// ---------------------------------------------------------------------------
// Host launcher
// ---------------------------------------------------------------------------
extern "C" void kernel_launch(void* const* d_in, const int* in_sizes, int n_in,
                              void* d_out, int out_size, void* d_ws, size_t ws_size,
                              hipStream_t stream) {
    (void)in_sizes; (void)n_in; (void)out_size; (void)ws_size;
    const int din[3]  = {1024, 2048, 2048};
    const int dout[3] = {2048, 2048, 1024};

    const float* x = (const float*)d_in[0];
    const float *W[3], *Bb[3], *Wa[3], *Ba[3], *We[3], *Be[3], *Wv[3], *Bv[3];
    for (int j = 0; j < 3; ++j) {
        const int base = 1 + 8 * j;
        W[j]  = (const float*)d_in[base + 0];
        Bb[j] = (const float*)d_in[base + 1];
        Wa[j] = (const float*)d_in[base + 2];
        Ba[j] = (const float*)d_in[base + 3];
        We[j] = (const float*)d_in[base + 4];
        Be[j] = (const float*)d_in[base + 5];
        Wv[j] = (const float*)d_in[base + 6];
        Bv[j] = (const float*)d_in[base + 7];
    }

    // carve workspace (deterministic layout, 256B-aligned)
    char* cur = (char*)d_ws;
    auto take = [&](size_t bytes) -> char* {
        char* r = cur;
        cur += (bytes + 255) & ~(size_t)255;
        return r;
    };
    __bf16 *Wm[3], *Weps[3], *Wab[3];
    float  *Blo[3], *Bhi[3];
    for (int j = 0; j < 3; ++j) {
        const size_t sz = (size_t)din[j] * dout[j];
        Wm[j]   = (__bf16*)take(sz * 2);
        Weps[j] = (__bf16*)take(sz * 2);
        Wab[j]  = (__bf16*)take(sz * 2);
        Blo[j]  = (float*)take((size_t)dout[j] * 4);
        Bhi[j]  = (float*)take((size_t)dout[j] * 4);
    }
    const size_t actb = (size_t)BATCH * 2048 * 2;  // bytes per bf16 activation matrix
    __bf16* A0mu = (__bf16*)take(actb);
    __bf16* A0ab = (__bf16*)take(actb);
    __bf16* A0r  = (__bf16*)take(actb);
    __bf16* A1mu = (__bf16*)take(actb);
    __bf16* A1ab = (__bf16*)take(actb);
    __bf16* A1r  = (__bf16*)take(actb);

    for (int j = 0; j < 3; ++j) {
        const int nw = din[j] * dout[j];
        prep_w_kernel<<<(nw + 255) / 256, 256, 0, stream>>>(
            W[j], Wa[j], We[j], Wv[j], Wm[j], Weps[j], Wab[j], nw);
        prep_b_kernel<<<(dout[j] + 255) / 256, 256, 0, stream>>>(
            Bb[j], Ba[j], Be[j], Bv[j], Blo[j], Bhi[j], dout[j]);
    }
    prep_x_kernel<<<(BATCH * 1024 + 255) / 256, 256, 0, stream>>>(
        x, A0mu, A0ab, A0r, BATCH * 1024);

    // layer 0: [2048,1024] -> [2048,2048], ReLU, emit bf16 operands
    ival_gemm_kernel<<<dim3(2048 / 128, BATCH / 128), 256, 0, stream>>>(
        A0mu, A0ab, A0r, Wm[0], Weps[0], Wab[0], Blo[0], Bhi[0],
        A1mu, A1ab, A1r, nullptr, BATCH, 2048, 1024, 1);
    // layer 1: [2048,2048] -> [2048,2048], ReLU
    ival_gemm_kernel<<<dim3(2048 / 128, BATCH / 128), 256, 0, stream>>>(
        A1mu, A1ab, A1r, Wm[1], Weps[1], Wab[1], Blo[1], Bhi[1],
        A0mu, A0ab, A0r, nullptr, BATCH, 2048, 2048, 1);
    // layer 2: [2048,2048] -> [2048,1024], no ReLU, interleaved fp32 [B,out,2]
    ival_gemm_kernel<<<dim3(1024 / 128, BATCH / 128), 256, 0, stream>>>(
        A0mu, A0ab, A0r, Wm[2], Weps[2], Wab[2], Blo[2], Bhi[2],
        nullptr, nullptr, nullptr, (float*)d_out, BATCH, 1024, 2048, 0);
}
